// Model_47012712022395
// MI455X (gfx1250) — matrix operand order repaired and checked by reading, not compiled, not run on hardware
//
#include <hip/hip_runtime.h>

typedef __attribute__((ext_vector_type(16))) _Float16 v16h;
typedef __attribute__((ext_vector_type(8)))  _Float16 v8h;
typedef __attribute__((ext_vector_type(8)))  float    v8f;
typedef __attribute__((ext_vector_type(4)))  float    v4f;

namespace {
constexpr int kT  = 300;
constexpr int kB  = 128;
constexpr int kNI = 128;
constexpr int kH  = 512;
constexpr int kNO = 32;

constexpr int kNWG  = 16;           // persistent workgroups
constexpr int kHC   = kH / kNWG;    // 32 columns of H per WG
constexpr int kNT   = kHC / 16;     // 2 N-tiles per WG
constexpr int kKTOT = kNI + kH;     // 640 combined K ([x | h_post])
constexpr int kKB   = kKTOT / 32;   // 20 K-blocks of 32
constexpr int kKBO  = kH / 32;      // 16 K-blocks for readout
constexpr int kNTO  = kNO / 16;     // 2 N-tiles for readout

constexpr float kAlpha  = 0.1f;
constexpr float kDtTauV = 0.1f;
constexpr float kDtSec  = 0.01f;
constexpr float kGamma  = 0.005f;

constexpr int kLdsWHalves  = kKB  * kNT  * 32 * 16;  // 20480 halves = 40 KB
constexpr int kLdsWoHalves = kKBO * kNTO * 32 * 16;  // 16384 halves = 32 KB
constexpr size_t kLdsBytes = (size_t)(kLdsWHalves + kLdsWoHalves) * 2; // 72 KB
} // namespace

// Grid-wide barrier: monotonically increasing counter, agent scope.
__device__ __forceinline__ void grid_barrier(unsigned* cnt, unsigned* epoch) {
  __syncthreads();
  if (threadIdx.x == 0) {
    __threadfence();  // release: publish our global stores
    __hip_atomic_fetch_add(cnt, 1u, __ATOMIC_RELAXED, __HIP_MEMORY_SCOPE_AGENT);
    const unsigned tgt = (++(*epoch)) * (unsigned)kNWG;
    while (__hip_atomic_load(cnt, __ATOMIC_RELAXED, __HIP_MEMORY_SCOPE_AGENT) < tgt) {
      __builtin_amdgcn_s_sleep(2);
    }
    __threadfence();  // acquire: invalidate stale near caches
  }
  __syncthreads();
}

__global__ __launch_bounds__(256, 1)
void rnn_stp_persistent(const float* __restrict__ x,       // (T,B,NI)
                        const float* __restrict__ h0,      // (B,H)
                        const float* __restrict__ sx0,     // (B,H)
                        const float* __restrict__ su0,     // (B,H)
                        const float* __restrict__ w_in,    // (NI,H)
                        const float* __restrict__ w_rnn,   // (H,H)
                        const float* __restrict__ b_rnn,   // (H)
                        const float* __restrict__ w_out,   // (H,NO)
                        const float* __restrict__ b_out,   // (NO)
                        const float* __restrict__ a_std,   // (H)
                        const float* __restrict__ a_stf,   // (H)
                        const float* __restrict__ Uv,      // (H)
                        const float* __restrict__ dynv,    // (H)
                        const float* __restrict__ mv,      // (H)
                        float* __restrict__ ys,            // (T,B,NO)
                        float* __restrict__ hs,            // (T,B,H)
                        float* __restrict__ ths,           // (T,B,H)
                        unsigned* __restrict__ cnt,
                        float* __restrict__ ws_sx,         // (B,H)
                        float* __restrict__ ws_su,         // (B,H)
                        _Float16* __restrict__ hpost,      // (B,H) f16
                        _Float16* __restrict__ h16)        // (B,H) f16
{
  extern __shared__ char smem_raw[];
  _Float16* lds_w  = (_Float16*)smem_raw;              // swizzled [w_in;w_rnn] slice
  _Float16* lds_wo = lds_w + kLdsWHalves;              // swizzled w_out (full)

  const int tid     = threadIdx.x;
  const int wg      = blockIdx.x;
  const int lane    = tid & 31;
  const int wv      = tid >> 5;
  const int g       = lane >> 4;    // lane group (K half selector)
  const int ln      = lane & 15;
  const int colbase = wg * kHC;

  // ---- one-time: swizzle weights (f32 -> f16) into B-fragment layout ----
  // fragment f = kb*kNT+nt; within fragment: lane L holds halves j=0..15 =
  // W[kb*32 + (j<8 ? (L/16)*8 + j : 16 + (L/16)*8 + j-8)][colbase + nt*16 + L%16]
  for (int p = tid; p < kKB * kNT * 32; p += 256) {
    const int frag = p >> 5, L = p & 31;
    const int kb = frag / kNT, nt = frag % kNT;
    const int gg = L >> 4, nn = L & 15;
    const int col = colbase + nt * 16 + nn;
    _Float16* dst = &lds_w[(size_t)p * 16];
    #pragma unroll
    for (int j = 0; j < 16; ++j) {
      const int k = kb * 32 + (j < 8 ? gg * 8 + j : 16 + gg * 8 + (j - 8));
      const float v = (k < kNI) ? w_in[(size_t)k * kH + col]
                                : w_rnn[(size_t)(k - kNI) * kH + col];
      dst[j] = (_Float16)v;
    }
  }
  for (int p = tid; p < kKBO * kNTO * 32; p += 256) {
    const int frag = p >> 5, L = p & 31;
    const int kb = frag / kNTO, nt = frag % kNTO;
    const int gg = L >> 4, nn = L & 15;
    const int col = nt * 16 + nn;
    _Float16* dst = &lds_wo[(size_t)p * 16];
    #pragma unroll
    for (int j = 0; j < 16; ++j) {
      const int k = kb * 32 + (j < 8 ? gg * 8 + j : 16 + gg * 8 + (j - 8));
      dst[j] = (_Float16)w_out[(size_t)k * kNO + col];
    }
  }
  __syncthreads();

  unsigned epoch = 0;

  for (int t = 0; t < kT; ++t) {
    const float* hprev = (t == 0) ? h0 : &hs[(size_t)(t - 1) * kB * kH];

    // ---------- Phase A: STP elementwise, publish h_post (own columns) ----------
    for (int e = tid; e < kB * kHC; e += 256) {
      const int b = e / kHC, c = e % kHC;
      const int col = colbase + c;
      const size_t idx = (size_t)b * kH + col;
      const float h  = hprev[idx];
      const float sx = (t == 0) ? sx0[idx] : ws_sx[idx];
      const float su = (t == 0) ? su0[idx] : ws_su[idx];
      const float d = dynv[col], u = Uv[col];
      float sx_n = sx + (a_std[col] * (1.f - sx) - kDtSec * su * sx * h) * d;
      float su_n = su + (a_stf[col] * (u - su) + kDtSec * u * (1.f - su) * h) * d;
      sx_n = fminf(1.f, fmaxf(0.f, sx_n));
      su_n = fminf(1.f, fmaxf(0.f, su_n));
      ws_sx[idx] = sx_n;
      ws_su[idx] = su_n;
      hpost[idx] = (_Float16)(su_n * sx_n * h);
    }
    // prefetch next timestep's input while we are at it
    if (t + 1 < kT) {
      const float* xn = &x[(size_t)(t + 1) * kB * kNI];
      __builtin_prefetch(&xn[(size_t)tid * 64], 0, 3);
    }
    grid_barrier(cnt, &epoch);  // h_post visible to all WGs

    // ---------- Phase B: z = [x_t | h_post] @ [w_in; w_rnn] (own columns) ----------
    {
      const int m0 = wv * 16;
      const int mrow = m0 + ln;           // A row for this lane
      const float* xt = &x[(size_t)t * kB * kNI];
      v8f acc[kNT] = {};
      for (int kb = 0; kb < kKB; ++kb) {
        const int k0 = kb * 32 + g * 8;
        v16h a;
        if (kb < kNI / 32) {  // x_t region: load f32, convert to f16
          const v4f* p0 = (const v4f*)&xt[(size_t)mrow * kNI + k0];
          const v4f f0 = p0[0], f1 = p0[1];
          const v4f* p1 = (const v4f*)&xt[(size_t)mrow * kNI + k0 + 16];
          const v4f f2 = p1[0], f3 = p1[1];
          #pragma unroll
          for (int j = 0; j < 4; ++j) {
            a[j]      = (_Float16)f0[j];
            a[4 + j]  = (_Float16)f1[j];
            a[8 + j]  = (_Float16)f2[j];
            a[12 + j] = (_Float16)f3[j];
          }
        } else {              // h_post region: already f16, two 16B loads
          const int kh = k0 - kNI;
          const v8h lo = *(const v8h*)&hpost[(size_t)mrow * kH + kh];
          const v8h hi = *(const v8h*)&hpost[(size_t)mrow * kH + kh + 16];
          #pragma unroll
          for (int j = 0; j < 8; ++j) { a[j] = lo[j]; a[8 + j] = hi[j]; }
        }
        #pragma unroll
        for (int nt = 0; nt < kNT; ++nt) {
          const v16h bfrag =
              *(const v16h*)&lds_w[((size_t)(kb * kNT + nt) * 32 + lane) * 16];
          acc[nt] = __builtin_amdgcn_wmma_f32_16x16x32_f16(
              false, a, false, bfrag, (short)0, acc[nt], false, false);
        }
      }
      // fused epilogue on the C layout: VGPR r -> row m0 + r + 8*g, lane%16 -> col
      const float* thprev = (t == 0) ? nullptr : &ths[(size_t)(t - 1) * kB * kH];
      float* hs_t = &hs[(size_t)t * kB * kH];
      float* th_t = &ths[(size_t)t * kB * kH];
      #pragma unroll
      for (int nt = 0; nt < kNT; ++nt) {
        const int col = colbase + nt * 16 + ln;
        const float br = b_rnn[col];
        const float mm = mv[col];
        #pragma unroll
        for (int r = 0; r < 8; ++r) {
          const int b = m0 + r + 8 * g;
          const size_t idx = (size_t)b * kH + col;
          const float th_p = (t == 0) ? 0.f : thprev[idx];
          const float h_p  = hprev[idx];
          const float z    = acc[nt][r] + br - kGamma * th_p;
          const float hn   = (1.f - kAlpha) * h_p + kAlpha * fmaxf(z, 0.f);
          const float thn  = (1.f - kDtTauV) * th_p + kDtTauV * fmaxf(mm * hn, 0.f);
          const float hc   = fminf(fmaxf(hn, 0.f), 100.f);
          th_t[idx] = thn;
          hs_t[idx] = hc;
          h16[idx]  = (_Float16)hc;
        }
      }
    }
    grid_barrier(cnt, &epoch);  // h_new (f16) visible to all WGs

    // ---------- Phase C: y = relu(h @ w_out + b_out); WGs 0..7, 2 waves each ----------
    if (wg < kB / 16 && wv < kNTO) {
      const int nt = wv;
      const int m0 = wg * 16;
      const int mrow = m0 + ln;
      v8f acc = {};
      for (int kb = 0; kb < kKBO; ++kb) {
        const int k0 = kb * 32 + g * 8;
        const v8h lo = *(const v8h*)&h16[(size_t)mrow * kH + k0];
        const v8h hi = *(const v8h*)&h16[(size_t)mrow * kH + k0 + 16];
        v16h a;
        #pragma unroll
        for (int j = 0; j < 8; ++j) { a[j] = lo[j]; a[8 + j] = hi[j]; }
        const v16h bfrag =
            *(const v16h*)&lds_wo[((size_t)(kb * kNTO + nt) * 32 + lane) * 16];
        acc = __builtin_amdgcn_wmma_f32_16x16x32_f16(
            false, a, false, bfrag, (short)0, acc, false, false);
      }
      const int col = nt * 16 + ln;
      const float bo = b_out[col];
      float* ys_t = &ys[(size_t)t * kB * kNO];
      #pragma unroll
      for (int r = 0; r < 8; ++r) {
        const int b = m0 + r + 8 * g;
        ys_t[(size_t)b * kNO + col] = fmaxf(acc[r] + bo, 0.f);
      }
    }
    // next iteration's barrier 1 orders Phase C(t) reads before Phase B(t+1) writes
  }
}

extern "C" void kernel_launch(void* const* d_in, const int* in_sizes, int n_in,
                              void* d_out, int out_size, void* d_ws, size_t ws_size,
                              hipStream_t stream) {
  (void)in_sizes; (void)n_in; (void)out_size; (void)ws_size;
  const float* x     = (const float*)d_in[0];
  const float* h0    = (const float*)d_in[1];
  const float* sx0   = (const float*)d_in[2];
  const float* su0   = (const float*)d_in[3];
  const float* w_in  = (const float*)d_in[4];
  const float* w_rnn = (const float*)d_in[5];
  const float* b_rnn = (const float*)d_in[6];
  const float* w_out = (const float*)d_in[7];
  const float* b_out = (const float*)d_in[8];
  const float* a_std = (const float*)d_in[9];
  const float* a_stf = (const float*)d_in[10];
  const float* Uv    = (const float*)d_in[11];
  const float* dynv  = (const float*)d_in[12];
  const float* mv    = (const float*)d_in[13];

  float* ys  = (float*)d_out;                        // (T,B,NO)
  float* hs  = ys + (size_t)kT * kB * kNO;           // (T,B,H)
  float* ths = hs + (size_t)kT * kB * kH;            // (T,B,H)

  char* ws = (char*)d_ws;
  unsigned* cnt   = (unsigned*)ws;                    // 256 B reserved
  float* ws_sx    = (float*)(ws + 256);
  float* ws_su    = ws_sx + (size_t)kB * kH;
  _Float16* hpost = (_Float16*)(ws_su + (size_t)kB * kH);
  _Float16* h16   = hpost + (size_t)kB * kH;

  // barrier counter must start at 0 every launch (capture-legal memset node)
  hipMemsetAsync(cnt, 0, 256, stream);

  rnn_stp_persistent<<<kNWG, 256, kLdsBytes, stream>>>(
      x, h0, sx0, su0, w_in, w_rnn, b_rnn, w_out, b_out,
      a_std, a_stf, Uv, dynv, mv,
      ys, hs, ths, cnt, ws_sx, ws_su, hpost, h16);
}
